// MMTLayer_88390426951924
// MI455X (gfx1250) — matrix operand order repaired and checked by reading, not compiled, not run on hardware
//
#include <hip/hip_runtime.h>
#include <hip/hip_bf16.h>

typedef __attribute__((ext_vector_type(16))) __bf16 v16bf;
typedef __attribute__((ext_vector_type(8)))  float  v8f;

#define TOK      4096      // B*S
#define SEQ      512
#define NBATCH   8
#define R        64
#define PCOLS    256       // 4 stacked projection matrices x 64
#define KTILE    32
#define LDPAD    36        // KTILE + 4 bf16 pad: 72B row stride -> 8B-aligned groups, conflict-free

// ---- WMMA fragment loaders (16-bit A 16x32 / B 32x16 per ISA layout tables) ----
__device__ __forceinline__ v16bf load_frag_a(const __bf16* lds, int lane) {
    // lanes 0-15: row=lane, K in {0..7, 16..23}; lanes 16-31: row=lane-16, K in {8..15, 24..31}
    const int row   = lane & 15;
    const int kbase = (lane < 16) ? 0 : 8;
    const __bf16* p = lds + row * LDPAD + kbase;
    v16bf f;
#pragma unroll
    for (int e = 0; e < 8; ++e)  f[e] = p[e];
#pragma unroll
    for (int e = 8; e < 16; ++e) f[e] = p[e + 8];
    return f;
}

__device__ __forceinline__ v16bf load_frag_b(const __bf16* lds, int lane) {
    // B is 32x16 (KxN) stored as W rows [n][k]; lanes 0-15: n=lane, K in {0..7, 16..23}; lanes 16-31: K in {8..15, 24..31}
    const int n     = lane & 15;
    const int kbase = (lane < 16) ? 0 : 8;
    const __bf16* p = lds + n * LDPAD + kbase;
    v16bf f;
#pragma unroll
    for (int e = 0; e < 8; ++e)  f[e] = p[e];
#pragma unroll
    for (int e = 8; e < 16; ++e) f[e] = p[e + 8];
    return f;
}

__device__ __forceinline__ v8f wmma_bf16(v16bf a, v16bf b, v8f c) {
    return __builtin_amdgcn_wmma_f32_16x16x32_bf16(false, a, false, b, (short)0, c, false, false);
}

// Two pipelined 4-WMMA bursts: DSread(A+B0..3) -> WMMA x4 (B4..7 loads issue
// underneath) -> WMMA x4. Partial dscnt waits, no WAR nops, low fragment liveness.
__device__ __forceinline__ void mma_tile(const __bf16* __restrict__ sXt,
                                         const __bf16* __restrict__ sWt,
                                         int wrow, int wcol, int lane, v8f (&acc)[8]) {
    v16bf fa = load_frag_a(sXt + wrow * LDPAD, lane);
    v16bf fb[8];
#pragma unroll
    for (int c = 0; c < 4; ++c) fb[c] = load_frag_b(sWt + (wcol + c * 16) * LDPAD, lane);
#pragma unroll
    for (int c = 0; c < 4; ++c) acc[c] = wmma_bf16(fa, fb[c], acc[c]);
#pragma unroll
    for (int c = 4; c < 8; ++c) fb[c] = load_frag_b(sWt + (wcol + c * 16) * LDPAD, lane);
#pragma unroll
    for (int c = 4; c < 8; ++c) acc[c] = wmma_bf16(fa, fb[c], acc[c]);
    __builtin_amdgcn_sched_group_barrier(0x100, 10, 0);  // A + B0..3 DS reads
    __builtin_amdgcn_sched_group_barrier(0x008,  4, 0);  // WMMA burst 1
    __builtin_amdgcn_sched_group_barrier(0x100,  8, 0);  // B4..7 DS reads (overlap burst 1)
    __builtin_amdgcn_sched_group_barrier(0x008,  4, 0);  // WMMA burst 2
}

__device__ __forceinline__ void lds_store8(__bf16* p, uint4 v) {   // 8 bf16, 8B-aligned halves
    uint2* q = (uint2*)p;
    q[0] = make_uint2(v.x, v.y);
    q[1] = make_uint2(v.z, v.w);
}

// ---- Kernel 0: fp32 -> bf16 conversion (one-time staging pass) ----
__global__ void cvt_kernel(const float* __restrict__ src, __bf16* __restrict__ dst, int n) {
    int i = blockIdx.x * blockDim.x + threadIdx.x;
    if (i < n) dst[i] = (__bf16)src[i];
}

// ---- Kernel 1: stacked projection GEMM  P[4096,256] = X[4096,D] @ Wstack^T ----
template<int D>
__global__ void __launch_bounds__(256, 1)
proj_kernel(const __bf16* __restrict__ Xb, const __bf16* __restrict__ Wb,
            float* __restrict__ P) {
    __shared__ __bf16 sX[2][64 * LDPAD];
    __shared__ __bf16 sW[2][PCOLS * LDPAD];

    const int tid  = threadIdx.x;          // 256 threads = 8 waves
    const int wave = tid >> 5, lane = tid & 31;
    const int row0 = blockIdx.x * 64;
    const int wrow = (wave >> 1) * 16;
    const int wcol = (wave & 1) * 128;

    const int xr = tid >> 2, xs = (tid & 3) * 8;   // X tile: 64 rows x 32k, 8 bf16/thread
    const int wn = tid;                            // W tile: 256 rows x 32k, 32 bf16/thread

    uint4 rx, rw[4];
    auto ldg = [&](int k0) {
        rx = *(const uint4*)(Xb + (size_t)(row0 + xr) * D + k0 + xs);
#pragma unroll
        for (int j = 0; j < 4; ++j)
            rw[j] = *(const uint4*)(Wb + (size_t)wn * D + k0 + j * 8);
    };
    auto stg = [&](int b) {
        lds_store8(&sX[b][xr * LDPAD + xs], rx);
#pragma unroll
        for (int j = 0; j < 4; ++j)
            lds_store8(&sW[b][wn * LDPAD + j * 8], rw[j]);
    };

    v8f acc[8];
#pragma unroll
    for (int c = 0; c < 8; ++c) acc[c] = v8f{0.f,0.f,0.f,0.f,0.f,0.f,0.f,0.f};

    const int nk = D / KTILE;
    ldg(0); stg(0); __syncthreads();
    for (int it = 0; it < nk; ++it) {
        if (it + 1 < nk) ldg((it + 1) * KTILE);        // global loads overlap with WMMA below
        mma_tile(sX[it & 1], sW[it & 1], wrow, wcol, lane, acc);
        if (it + 1 < nk) stg((it + 1) & 1);            // other buffer: no hazard with reads
        __syncthreads();
    }

    const int crow = row0 + wrow + ((lane < 16) ? 0 : 8);
    const int ccol = wcol + (lane & 15);
#pragma unroll
    for (int c = 0; c < 8; ++c)
#pragma unroll
        for (int e = 0; e < 8; ++e)
            P[(size_t)(crow + e) * PCOLS + ccol + c * 16] = acc[c][e];
}

// ---- Kernel 2: rank-1 factors  a = P[:,0:64]*P[:,64:128], c = P[:,128:192]*P[:,192:256] ----
__global__ void ac_kernel(const float* __restrict__ Pall,
                          float* __restrict__ Aall, float* __restrict__ Call) {
    int idx = blockIdx.x * blockDim.x + threadIdx.x;   // 3 * 4096 * 64
    int mod = idx >> 18;
    int r   = idx & 262143;
    int s = r >> 6, j = r & 63;
    const float* P = Pall + (size_t)mod * (TOK * PCOLS) + (size_t)s * PCOLS;
    Aall[idx] = P[j]       * P[64 + j];
    Call[idx] = P[128 + j] * P[192 + j];
}

// ---- Kernel 3: per-batch second moment  M[b] = (1/S) * A_b^T C_b  (64x64) ----
__global__ void m_kernel(const float* __restrict__ Aall, const float* __restrict__ Call,
                         float* __restrict__ Mall) {
    const int mod = blockIdx.x >> 3, b = blockIdx.x & 7;
    const float* A = Aall + (size_t)mod * (TOK * R) + (size_t)b * SEQ * R;
    const float* C = Call + (size_t)mod * (TOK * R) + (size_t)b * SEQ * R;
    __shared__ float sa[R], sc[R];
    const int t  = threadIdx.x;
    const int x  = t >> 2;
    const int y0 = (t & 3) * 16;
    float acc[16];
#pragma unroll
    for (int e = 0; e < 16; ++e) acc[e] = 0.f;
    for (int s = 0; s < SEQ; ++s) {
        if (t < 64)        sa[t]      = A[s * R + t];
        else if (t < 128)  sc[t - 64] = C[s * R + (t - 64)];
        __syncthreads();
        float av = sa[x];
#pragma unroll
        for (int e = 0; e < 16; ++e) acc[e] += av * sc[y0 + e];
        __syncthreads();
    }
    float* M = Mall + (size_t)(mod * NBATCH + b) * (R * R);
#pragma unroll
    for (int e = 0; e < 16; ++e) M[x * R + y0 + e] = acc[e] * (1.0f / (float)SEQ);
}

// ---- Kernel 4: fused E = M_first @ M_second ; D[s,:] = C[s,:] @ E ----
__global__ void ed_kernel(const float* __restrict__ Mall, const float* __restrict__ Call,
                          float* __restrict__ Dall) {
    const int mod = blockIdx.x >> 3, b = blockIdx.x & 7;
    const int fm = (mod == 0) ? 2 : (mod == 1) ? 0 : 1;   // text:Ma@Mv video:Mt@Ma audio:Mv@Mt
    const int sm = (mod == 0) ? 1 : (mod == 1) ? 2 : 0;
    __shared__ float sF[R * R], sS[R * R], sE[R * R];
    const float* MF = Mall + (size_t)(fm * NBATCH + b) * (R * R);
    const float* MS = Mall + (size_t)(sm * NBATCH + b) * (R * R);
    const int t = threadIdx.x;
    for (int i = t; i < R * R; i += 256) { sF[i] = MF[i]; sS[i] = MS[i]; }
    __syncthreads();
    for (int o = t; o < R * R; o += 256) {
        int i = o >> 6, k = o & 63;
        float acc = 0.f;
#pragma unroll 8
        for (int j = 0; j < R; ++j) acc += sF[i * R + j] * sS[j * R + k];
        sE[o] = acc;
    }
    __syncthreads();
    const float* C  = Call + (size_t)mod * (TOK * R) + (size_t)b * SEQ * R;
    float*       Dd = Dall + (size_t)mod * (TOK * R) + (size_t)b * SEQ * R;
    for (int o = t; o < SEQ * R; o += 256) {
        int s = o >> 6, w = o & 63;
        const float* c = C + s * R;
        float acc = 0.f;
#pragma unroll 8
        for (int z = 0; z < R; ++z) acc += c[z] * sE[z * R + w];
        Dd[o] = acc;
    }
}

// ---- Kernel 5: output GEMM  Y = X * ((kron(a,d) @ W^T) + bias + 1) ----
template<int NDIM>
__global__ void __launch_bounds__(256, 1)
out_kernel(const float* __restrict__ A, const float* __restrict__ Dd,
           const __bf16* __restrict__ Wb, const float* __restrict__ bias,
           const float* __restrict__ X, float* __restrict__ Y) {
    __shared__ __bf16 sK[2][64 * LDPAD];
    __shared__ __bf16 sW[2][PCOLS * LDPAD];

    const int tid  = threadIdx.x;
    const int wave = tid >> 5, lane = tid & 31;
    const int row0 = blockIdx.x * 64;
    const int col0 = blockIdx.y * 256;
    const int wrow = (wave >> 1) * 16;
    const int wcol = (wave & 1) * 128;

    const int kr = tid >> 2, ks = (tid & 3) * 8;   // kron tile: 8 elems/thread
    const int wn = tid;                            // W tile: 32 bf16/thread

    float  ra;
    float4 rd0, rd1;
    uint4  rw[4];
    auto ldg = [&](int k0) {
        const int gr = row0 + kr;
        ra  = A[(size_t)gr * R + (k0 >> 6)];
        const float4* dp = (const float4*)(Dd + (size_t)gr * R + (k0 & 63) + ks);
        rd0 = dp[0]; rd1 = dp[1];
#pragma unroll
        for (int j = 0; j < 4; ++j)
            rw[j] = *(const uint4*)(Wb + (size_t)(col0 + wn) * (R * R) + k0 + j * 8);
        if (k0 + KTILE < R * R)   // pull next K-tile toward WGP (global_prefetch_b8)
            __builtin_prefetch(Wb + (size_t)(col0 + wn) * (R * R) + k0 + KTILE, 0, 1);
    };
    auto stg = [&](int b) {
        __bf16* p = &sK[b][kr * LDPAD + ks];
        p[0] = (__bf16)(ra * rd0.x); p[1] = (__bf16)(ra * rd0.y);
        p[2] = (__bf16)(ra * rd0.z); p[3] = (__bf16)(ra * rd0.w);
        p[4] = (__bf16)(ra * rd1.x); p[5] = (__bf16)(ra * rd1.y);
        p[6] = (__bf16)(ra * rd1.z); p[7] = (__bf16)(ra * rd1.w);
#pragma unroll
        for (int j = 0; j < 4; ++j)
            lds_store8(&sW[b][wn * LDPAD + j * 8], rw[j]);
    };

    v8f acc[8];
#pragma unroll
    for (int c = 0; c < 8; ++c) acc[c] = v8f{0.f,0.f,0.f,0.f,0.f,0.f,0.f,0.f};

    const int nk = (R * R) / KTILE;   // 128
    ldg(0); stg(0); __syncthreads();
    for (int it = 0; it < nk; ++it) {
        if (it + 1 < nk) ldg((it + 1) * KTILE);
        mma_tile(sK[it & 1], sW[it & 1], wrow, wcol, lane, acc);
        if (it + 1 < nk) stg((it + 1) & 1);
        __syncthreads();
    }

    const int crow = row0 + wrow + ((lane < 16) ? 0 : 8);
    const int ccol = col0 + wcol + (lane & 15);
#pragma unroll
    for (int c = 0; c < 8; ++c) {
        int cc = ccol + c * 16;
        float bv = bias[cc] + 1.0f;    // fold residual: y = x*(attn + b + 1)
#pragma unroll
        for (int e = 0; e < 8; ++e) {
            int rr = crow + e;
            float xv = X[(size_t)rr * NDIM + cc];
            Y[(size_t)rr * NDIM + cc] = xv * (acc[c][e] + bv);
        }
    }
}

extern "C" void kernel_launch(void* const* d_in, const int* in_sizes, int n_in,
                              void* d_out, int out_size, void* d_ws, size_t ws_size,
                              hipStream_t stream) {
    const float* t   = (const float*)d_in[0];
    const float* v   = (const float*)d_in[1];
    const float* a   = (const float*)d_in[2];
    const float* Qt1 = (const float*)d_in[3];
    const float* Qt2 = (const float*)d_in[4];
    const float* Qv1 = (const float*)d_in[5];
    const float* Qv2 = (const float*)d_in[6];
    const float* Qa1 = (const float*)d_in[7];
    const float* Qa2 = (const float*)d_in[8];
    const float* Kt1 = (const float*)d_in[9];
    const float* Kt2 = (const float*)d_in[10];
    const float* Kv1 = (const float*)d_in[11];
    const float* Kv2 = (const float*)d_in[12];
    const float* Ka1 = (const float*)d_in[13];
    const float* Ka2 = (const float*)d_in[14];
    const float* Wt  = (const float*)d_in[15];
    const float* bt  = (const float*)d_in[16];
    const float* Wv  = (const float*)d_in[17];
    const float* bv  = (const float*)d_in[18];
    const float* Wa  = (const float*)d_in[19];
    const float* ba  = (const float*)d_in[20];
    float* Y  = (float*)d_out;

    // ---- workspace layout ----
    float* P  = (float*)d_ws;                    // 3 * 4096*256
    float* A  = P  + 3 * TOK * PCOLS;            // 3 * 4096*64
    float* C  = A  + 3 * TOK * R;
    float* M  = C  + 3 * TOK * R;                // 3 * 8*64*64
    float* Dd = M  + 3 * NBATCH * R * R;         // 3 * 4096*64
    __bf16* bbase = (__bf16*)(Dd + 3 * TOK * R);
    __bf16* Xbt = bbase;                         // 4096*768
    __bf16* Xbv = Xbt + TOK * 768;               // 4096*512
    __bf16* Xba = Xbv + TOK * 512;
    __bf16* Wst = Xba + TOK * 512;               // 256*768 (Q1,K1,Q2,K2 stacked)
    __bf16* Wsv = Wst + PCOLS * 768;             // 256*512
    __bf16* Wsa = Wsv + PCOLS * 512;
    __bf16* Wot = Wsa + PCOLS * 512;             // 768*4096
    __bf16* Wov = Wot + 768 * (R * R);           // 512*4096
    __bf16* Woa = Wov + 512 * (R * R);

    dim3 blk(256);
    auto cvt = [&](const float* s, __bf16* d, int n) {
        cvt_kernel<<<dim3((n + 255) / 256), blk, 0, stream>>>(s, d, n);
    };

    // 0) bf16 staging pass
    cvt(t, Xbt, TOK * 768);  cvt(v, Xbv, TOK * 512);  cvt(a, Xba, TOK * 512);
    cvt(Qt1, Wst,           R * 768); cvt(Kt1, Wst +  64 * 768, R * 768);
    cvt(Qt2, Wst + 128*768, R * 768); cvt(Kt2, Wst + 192 * 768, R * 768);
    cvt(Qv1, Wsv,           R * 512); cvt(Kv1, Wsv +  64 * 512, R * 512);
    cvt(Qv2, Wsv + 128*512, R * 512); cvt(Kv2, Wsv + 192 * 512, R * 512);
    cvt(Qa1, Wsa,           R * 512); cvt(Ka1, Wsa +  64 * 512, R * 512);
    cvt(Qa2, Wsa + 128*512, R * 512); cvt(Ka2, Wsa + 192 * 512, R * 512);
    cvt(Wt, Wot, 768 * R * R);  cvt(Wv, Wov, 512 * R * R);  cvt(Wa, Woa, 512 * R * R);

    // 1) projections
    proj_kernel<768><<<dim3(TOK / 64), blk, 0, stream>>>(Xbt, Wst, P);
    proj_kernel<512><<<dim3(TOK / 64), blk, 0, stream>>>(Xbv, Wsv, P + TOK * PCOLS);
    proj_kernel<512><<<dim3(TOK / 64), blk, 0, stream>>>(Xba, Wsa, P + 2 * TOK * PCOLS);

    // 2) rank-1 factors a,c
    ac_kernel<<<dim3(3 * TOK * R / 256), blk, 0, stream>>>(P, A, C);

    // 3) per-batch moments M
    m_kernel<<<dim3(3 * NBATCH), blk, 0, stream>>>(A, C, M);

    // 4) E = M_i @ M_j ; D = C @ E
    ed_kernel<<<dim3(3 * NBATCH), blk, 0, stream>>>(M, C, Dd);

    // 5) output projections + fused residual epilogue
    out_kernel<768><<<dim3(TOK / 64, 3), blk, 0, stream>>>(A, Dd, Wot, bt, t, Y);
    out_kernel<512><<<dim3(TOK / 64, 2), blk, 0, stream>>>(A + TOK * R,     Dd + TOK * R,     Wov, bv, v, Y + NBATCH * SEQ * 768);
    out_kernel<512><<<dim3(TOK / 64, 2), blk, 0, stream>>>(A + 2 * TOK * R, Dd + 2 * TOK * R, Woa, ba, a, Y + NBATCH * SEQ * (768 + 512));
}
